// AutoEncoder_21715354648821
// MI455X (gfx1250) — matrix operand order repaired and checked by reading, not compiled, not run on hardware
//
#include <hip/hip_runtime.h>
#include <hip/hip_bf16.h>
#include <math.h>

#define BATCH 8192
#define EMBED 1024
#define NFEAT 32768
#define TOPK  64
#define ROWS_PER_BLOCK 16
#define NWAVES 8
#define NBLK   4     // N-tiles per wave per pass (4 accumulators)

typedef __attribute__((ext_vector_type(16))) __bf16 bf16x16;
typedef __attribute__((ext_vector_type(8)))  float  f32x8;
typedef __attribute__((ext_vector_type(4)))  float  f32x4;
typedef __attribute__((ext_vector_type(4)))  unsigned int u32x4;
typedef __attribute__((ext_vector_type(2)))  unsigned int u32x2;

union FragBF16 {
    bf16x16 v;
    u32x4   q[2];
};

// ---------------------------------------------------------------------------
// Kernel 1: fp32 -> bf16 conversion (round-to-nearest-even), vectorized x4.
// ---------------------------------------------------------------------------
__device__ __forceinline__ unsigned int pack_bf16_pair(float a, float b) {
    unsigned int ua = __float_as_uint(a);
    unsigned int ub = __float_as_uint(b);
    ua += 0x7FFFu + ((ua >> 16) & 1u);
    ub += 0x7FFFu + ((ub >> 16) & 1u);
    return (ua >> 16) | (ub & 0xFFFF0000u);
}

__global__ __launch_bounds__(256)
void f32_to_bf16_kernel(const f32x4* __restrict__ src, u32x2* __restrict__ dst, int n4) {
    int i = blockIdx.x * blockDim.x + threadIdx.x;
    const int stride = gridDim.x * blockDim.x;
    for (; i < n4; i += stride) {
        f32x4 v = src[i];
        u32x2 o;
        o.x = pack_bf16_pair(v.x, v.y);
        o.y = pack_bf16_pair(v.z, v.w);
        dst[i] = o;
    }
}

// ---------------------------------------------------------------------------
// Kernel 2: fused  sigmoid(embed @ enc_w^T + b)  -> top-64 -> weighted decode.
//
// Block: 256 threads (8 waves, wave32). Each block owns 16 batch rows.
// Wave w processes groups of NBLK=4 consecutive 16x16 output tiles:
// the A fragment is loaded once per K-step and feeds 4 WMMAs.
// ---------------------------------------------------------------------------
__global__ __launch_bounds__(256)
void sae_fused_kernel(const __bf16* __restrict__ embB,   // [BATCH][EMBED] bf16
                      const __bf16* __restrict__ wB,     // [NFEAT][EMBED] bf16
                      const float*  __restrict__ bias,   // [NFEAT]
                      const float*  __restrict__ dec,    // [NFEAT][EMBED] f32
                      float*        __restrict__ out)    // [BATCH][EMBED] f32
{
    __shared__ float          tileV[NWAVES][256];            //  8 KB
    __shared__ float          topV [NWAVES][16][TOPK];       // 32 KB
    __shared__ unsigned short topI [NWAVES][16][TOPK];       // 16 KB
    __shared__ float          finV [16][TOPK];               //  4 KB
    __shared__ unsigned short finI [16][TOPK];               //  2 KB

    const int tid  = threadIdx.x;
    const int wave = tid >> 5;
    const int lane = tid & 31;
    const int hi   = lane >> 4;     // lane half: 0 = lanes 0-15, 1 = lanes 16-31
    const int l16  = lane & 15;
    const int rowBase = blockIdx.x * ROWS_PER_BLOCK;

    // Initialize top-k lists to sentinel (sigmoid outputs are in (0,1)).
    for (int i = tid; i < NWAVES * 16 * TOPK; i += 256) {
        (&topV[0][0][0])[i] = -1e30f;
        (&topI[0][0][0])[i] = 0;
    }
    __syncthreads();

    // Per-lane running min of this lane's (row = l16) top-k list.
    float curMin  = -1e30f;
    int   curSlot = 0;

    // A fragment (16-bit A 16x32): element i<8 -> K = kk + hi*8 + i
    //                              element i>=8 -> K = kk + 16 + hi*8 + (i-8)
    const __bf16* aBase = embB + (size_t)(rowBase + l16) * EMBED + hi * 8;

    const int groupsPerWave = NFEAT / 16 / NWAVES / NBLK;   // 64

    for (int g = 0; g < groupsPerWave; ++g) {
        const int ntile0 = (g * NWAVES + wave) * (NBLK * 16);
        // B fragment (16-bit B 32x16): element i<8 -> K = kk + hi*8 + i, i>=8 -> K = kk + 16 + hi*8 + (i-8).
        // Tile j is a constant byte offset (j*16 rows * 2KB) -> folds into
        // the instruction immediate offset.
        const __bf16* bBase = wB + (size_t)(ntile0 + l16) * EMBED + hi * 8;

        f32x8 acc[NBLK];
        #pragma unroll
        for (int j = 0; j < NBLK; ++j)
            acc[j] = (f32x8){0.f, 0.f, 0.f, 0.f, 0.f, 0.f, 0.f, 0.f};

        #pragma unroll 2
        for (int kk = 0; kk < EMBED; kk += 32) {
            FragBF16 fa;
            fa.q[0] = *(const u32x4*)(aBase + kk);
            fa.q[1] = *(const u32x4*)(aBase + kk + 16);
            #pragma unroll
            for (int j = 0; j < NBLK; ++j) {
                FragBF16 fb;
                fb.q[0] = *(const u32x4*)(bBase + (size_t)j * 16 * EMBED + kk);
                fb.q[1] = *(const u32x4*)(bBase + (size_t)j * 16 * EMBED + kk + 16);
                acc[j] = __builtin_amdgcn_wmma_f32_16x16x32_bf16(
                    false, fa.v, false, fb.v, (short)0, acc[j], false, false);
            }
        }

        // Epilogue per tile: bias + fast sigmoid, stage to LDS, top-k insert.
        // Fully unrolled over j so acc[j][r] stays statically register-indexed
        // (avoids m0/v_movrels dynamic VGPR indexing).
        #pragma unroll
        for (int j = 0; j < NBLK; ++j) {
            const int ntile = ntile0 + j * 16;
            const float b = bias[ntile + l16];
            // C/D layout: lane column n = l16; VGPR r holds row M = r + hi*8.
            #pragma unroll
            for (int r = 0; r < 8; ++r) {
                float x = acc[j][r] + b;
                float s = __builtin_amdgcn_rcpf(1.0f + __expf(-x));
                tileV[wave][(r + hi * 8) * 16 + l16] = s;
            }

            // Lanes 0-15: lane owns row = l16; merge 16 candidates into top-64.
            if (hi == 0) {
                #pragma unroll 1
                for (int c = 0; c < 16; ++c) {
                    float v = tileV[wave][l16 * 16 + c];
                    if (v > curMin) {
                        topV[wave][l16][curSlot] = v;
                        topI[wave][l16][curSlot] = (unsigned short)(ntile + c);
                        float mn = topV[wave][l16][0];
                        int   ms = 0;
                        #pragma unroll 1
                        for (int k = 1; k < TOPK; ++k) {
                            float q = topV[wave][l16][k];
                            if (q < mn) { mn = q; ms = k; }
                        }
                        curMin = mn; curSlot = ms;
                    }
                }
            }
        }
    }
    __syncthreads();

    // Final merge: 512 candidates per row -> top-64. Wave w handles rows 2w,2w+1.
    for (int rr = 0; rr < 2; ++rr) {
        const int row = wave * 2 + rr;
        for (int sel = 0; sel < TOPK; ++sel) {
            float bv = -1e30f;
            int   bc = 0;
            for (int c = lane; c < NWAVES * TOPK; c += 32) {
                float v = topV[c >> 6][row][c & 63];
                if (v > bv) { bv = v; bc = c; }
            }
            #pragma unroll
            for (int off = 16; off >= 1; off >>= 1) {
                float ov = __shfl_xor(bv, off, 32);
                int   oc = __shfl_xor(bc, off, 32);
                if (ov > bv || (ov == bv && oc < bc)) { bv = ov; bc = oc; }
            }
            if (lane == 0) {
                finV[row][sel] = bv;
                finI[row][sel] = topI[bc >> 6][row][bc & 63];
                topV[bc >> 6][row][bc & 63] = -1e30f;   // consume
            }
        }
    }
    __syncthreads();

    // Decoder: out[row,:] = sum_k w_k * dec_table[f_k, :].  256 threads x float4.
    const f32x4* dec4 = (const f32x4*)dec;
    for (int row = 0; row < ROWS_PER_BLOCK; ++row) {
        f32x4 a = {0.f, 0.f, 0.f, 0.f};
        #pragma unroll 1
        for (int k = 0; k < TOPK; ++k) {
            float w = finV[row][k];
            int   f = finI[row][k];
            f32x4 d = dec4[(size_t)f * (EMBED / 4) + tid];
            a += w * d;
        }
        *(f32x4*)(out + (size_t)(rowBase + row) * EMBED + tid * 4) = a;
    }
}

// ---------------------------------------------------------------------------
// Host-side launcher.
// Inputs (setup_inputs order): embed f32[8192*1024], enc_w f32[32768*1024],
//                              enc_b f32[32768], dec_table f32[32768*1024].
// Output: f32[8192*1024].
// Workspace: embed_bf16 (16 MB) + enc_w_bf16 (64 MB) = 80 MB.
// ---------------------------------------------------------------------------
extern "C" void kernel_launch(void* const* d_in, const int* in_sizes, int n_in,
                              void* d_out, int out_size, void* d_ws, size_t ws_size,
                              hipStream_t stream) {
    const float* embed = (const float*)d_in[0];
    const float* enc_w = (const float*)d_in[1];
    const float* enc_b = (const float*)d_in[2];
    const float* dec   = (const float*)d_in[3];
    float* out = (float*)d_out;

    __bf16* embB = (__bf16*)d_ws;
    __bf16* wB   = embB + (size_t)BATCH * EMBED;

    f32_to_bf16_kernel<<<1024, 256, 0, stream>>>(
        (const f32x4*)embed, (u32x2*)embB, BATCH * EMBED / 4);
    f32_to_bf16_kernel<<<2048, 256, 0, stream>>>(
        (const f32x4*)enc_w, (u32x2*)wB, NFEAT * EMBED / 4);

    sae_fused_kernel<<<BATCH / ROWS_PER_BLOCK, 256, 0, stream>>>(
        embB, wB, enc_b, dec, out);
}
